// ModConv_77343771066996
// MI455X (gfx1250) — matrix operand order repaired and checked by reading, not compiled, not run on hardware
//
#include <hip/hip_runtime.h>

typedef __attribute__((ext_vector_type(16))) __bf16 v16bf;
typedef __attribute__((ext_vector_type(8)))  float  v8f;
typedef __attribute__((ext_vector_type(4)))  int    v4i;

typedef __attribute__((address_space(1))) v4i* gptr_v4i;   // global
typedef __attribute__((address_space(3))) v4i* lptr_v4i;   // LDS

constexpr int NB  = 16;     // batch
constexpr int CI  = 256;    // in channels
constexpr int CO  = 256;    // out channels
constexpr int SD  = 512;    // style dim
constexpr int HH  = 128;
constexpr int WW  = 128;
constexpr int HWP = HH * WW;      // 16384
constexpr int KT  = CI * 9;       // 2304  (GEMM K), ordered k = (kh*3+kw)*CI + ci
constexpr float SCALE = 0.02083333333f;  // 1/sqrt(Cin*K*K) = 1/48

#if __has_builtin(__builtin_amdgcn_global_load_async_to_lds_b128)
#define HAS_ASYNC_LDS 1
#else
#define HAS_ASYNC_LDS 0
#endif

#if __has_builtin(__builtin_amdgcn_s_wait_asynccnt)
#define WAIT_ASYNC() __builtin_amdgcn_s_wait_asynccnt(0)
#else
#define WAIT_ASYNC() asm volatile("s_wait_asynccnt 0x0" ::: "memory")
#endif

// ---- step 1a: wsum[co][ci] = scale^2 * sum_k w^2 ----------------------------
__global__ void wsum_kernel(const float* __restrict__ weight, float* __restrict__ wsum) {
    int idx = blockIdx.x * 256 + threadIdx.x;           // CO*CI = 65536
    if (idx >= CO * CI) return;
    const float* wp = weight + idx * 9;
    float acc = 0.f;
#pragma unroll
    for (int i = 0; i < 9; ++i) { float v = wp[i]; acc += v * v; }
    wsum[idx] = acc * SCALE * SCALE;
}

// ---- step 1b: s[b][ci] = style[b]·mod_w[ci] + mod_b[ci] ---------------------
__global__ void style_kernel(const float* __restrict__ style, const float* __restrict__ mod_w,
                             const float* __restrict__ mod_b, float* __restrict__ s) {
    int idx = blockIdx.x * 256 + threadIdx.x;           // NB*CI = 4096
    int b = idx / CI, ci = idx % CI;
    const float* st = style + b * SD;
    const float* mw = mod_w + ci * SD;
    float acc = 0.f;
    for (int k = 0; k < SD; ++k) acc += st[k] * mw[k];
    s[idx] = acc + mod_b[ci];
}

// ---- step 1c: demod[b][co] = rsqrt( sum_ci s^2 * wsum + eps ) ---------------
__global__ void demod_kernel(const float* __restrict__ s, const float* __restrict__ wsum,
                             float* __restrict__ demod) {
    int idx = blockIdx.x * 256 + threadIdx.x;           // NB*CO = 4096
    int b = idx / CO, co = idx % CO;
    const float* sb = s + b * CI;
    const float* wr = wsum + co * CI;
    float acc = 1e-8f;
    for (int c = 0; c < CI; ++c) { float sv = sb[c]; acc += sv * sv * wr[c]; }
    demod[idx] = rsqrtf(acc);
}

// ---- step 2: bf16 modulated weights, K ordered (kh,kw) outer / ci inner -----
__global__ void prep_kernel(const float* __restrict__ weight, const float* __restrict__ s,
                            const float* __restrict__ demod, __bf16* __restrict__ wb) {
    long idx = (long)blockIdx.x * 256 + threadIdx.x;    // NB*CO*KT = 9,437,184
    int  k2   = (int)(idx % KT);
    long bc   = idx / KT;
    int  co   = (int)(bc % CO);
    int  b    = (int)(bc / CO);
    int  kidx = k2 >> 8;          // kh*3+kw  (CI == 256)
    int  ci   = k2 & 255;
    float v = SCALE * weight[(co * CI + ci) * 9 + kidx] * s[b * CI + ci] * demod[b * CO + co];
    wb[idx] = (__bf16)v;
}

// ---- step 3: implicit GEMM conv, bf16 WMMA, double-buffered LDS pipeline ----
// grid: (H, CO/64, NB), block 256 (8 waves). Block tile: 64 cout x 128 pixels.
// Wave (2x4 grid) owns 32x32 output -> 4 accumulators, 4 WMMAs / K-step.
// A tile staged with GLOBAL_LOAD_ASYNC_TO_LDS_B128 when the builtin exists.
__global__ __launch_bounds__(256) void conv_kernel(const float* __restrict__ x,
                                                   const __bf16* __restrict__ wb,
                                                   const float* __restrict__ bias,
                                                   float* __restrict__ out) {
    __shared__ __bf16 As[2][64][32];     // [buf][m][k]  2 x 4 KB
    __shared__ __bf16 Bs[2][128][32];    // [buf][n][k]  2 x 8 KB

    const int h   = blockIdx.x;          // output row
    const int cb  = blockIdx.y * 64;     // cout base
    const int b   = blockIdx.z;

    const int tid  = threadIdx.x;
    const int wv   = tid >> 5;
    const int lane = tid & 31;

    // A staging: thread -> row m=tid/4, 8 bf16 at (tid%4)*8 (one 16B transfer)
    const int am = tid >> 2;
    const int ak = (tid & 3) * 8;
    const __bf16* wbRow = wb + ((long)(b * CO + cb + am)) * KT + ak;

    // B staging: thread -> pixel n=tid&127, 16 consecutive k(=ci) at (tid>>7)*16
    const int bn = tid & 127;
    const int bk = (tid >> 7) * 16;
    const float* xb = x + (long)b * CI * HWP;

    v8f c00 = {}, c01 = {}, c10 = {}, c11 = {};

    float bReg[16];
#if !HAS_ASYNC_LDS
    uint4 aReg;
#endif

    // ---- A tile staging ----
#if HAS_ASYNC_LDS
    auto stageA = [&](int kk, int q) {      // DMA straight into LDS, ASYNCcnt-tracked
        __builtin_amdgcn_global_load_async_to_lds_b128(
            (gptr_v4i)(wbRow + kk),
            (lptr_v4i)(&As[q][am][ak]),
            0, 0);
    };
#else
    auto loadA   = [&](int kk) { aReg = *(const uint4*)(wbRow + kk); };
    auto commitA = [&](int q)  { *(uint4*)(&As[q][am][ak]) = aReg; };
#endif

    // ---- B tile staging (needs f32 -> bf16 conversion, register path) ----
    auto loadB = [&](int kk) {
        int kidx = kk >> 8;               // uniform: kernel position
        int kh = kidx / 3;
        int kw = kidx - 3 * kh;
        int ih = h + kh - 1;
        int iw = bn + kw - 1;
        bool ok = ((unsigned)ih < (unsigned)HH) && ((unsigned)iw < (unsigned)WW);
        const float* p = xb + ((long)((kk & 255) + bk)) * HWP + ih * WW + iw;
#pragma unroll
        for (int j = 0; j < 16; ++j) bReg[j] = 0.f;
        if (ok) {
#pragma unroll
            for (int j = 0; j < 16; ++j) bReg[j] = p[(long)j * HWP];  // one clause
        }
    };
    auto commitB = [&](int q) {
        __bf16 tmp[16];
#pragma unroll
        for (int j = 0; j < 16; ++j) tmp[j] = (__bf16)bReg[j];
        *(uint4*)(&Bs[q][bn][bk])     = *(const uint4*)(&tmp[0]);
        *(uint4*)(&Bs[q][bn][bk + 8]) = *(const uint4*)(&tmp[8]);
    };

    // ---- prologue: stage tile 0 into buffer 0 ----
#if HAS_ASYNC_LDS
    stageA(0, 0);
#else
    loadA(0);
#endif
    loadB(0);
    commitB(0);
#if HAS_ASYNC_LDS
    WAIT_ASYNC();                        // A DMA done before barrier signal
#else
    commitA(0);
#endif

    // fragment indices (K contiguous per lane half, same for A and B)
    const int mb   = ((wv >> 2) << 5);          // wave M base: 0 or 32
    const int nb_  = ((wv & 3) << 5);           // wave N base: 0..96
    const int mrow = lane & 15;
    const int koff = (lane >> 4) << 3;          // unused (A and B both index K via half)
    const int half = (lane >> 4) << 4;          // B: K halves {0,16}

    int p = 0;
    for (int kk = 0; kk < KT; kk += 32) {
        __syncthreads();                        // buf p now visible to all waves
        const bool more = (kk + 32) < KT;
        if (more) {
#if HAS_ASYNC_LDS
            stageA(kk + 32, p ^ 1);             // DMA lands during WMMAs
#else
            loadA(kk + 32);
#endif
            loadB(kk + 32);                     // global clause flies over WMMAs
        }

        // --- A fragments: two 16B chunks per lane (K=half..half+15, contiguous) ---
        v16bf a0, a1, b0, b1;
        *(uint4*)(&a0)     = *(const uint4*)(&As[p][mb + mrow][half]);
        *((uint4*)(&a0)+1) = *(const uint4*)(&As[p][mb + mrow][half + 8]);
        *(uint4*)(&a1)     = *(const uint4*)(&As[p][mb + mrow + 16][half]);
        *((uint4*)(&a1)+1) = *(const uint4*)(&As[p][mb + mrow + 16][half + 8]);
        // --- B fragments: contiguous 32B per lane (K=half..half+15, N=lane&15) ---
        b0 = *(const v16bf*)(&Bs[p][nb_ + (lane & 15)][half]);
        b1 = *(const v16bf*)(&Bs[p][nb_ + (lane & 15) + 16][half]);

        c00 = __builtin_amdgcn_wmma_f32_16x16x32_bf16(false, a0, false, b0, (short)0, c00, false, false);
        c01 = __builtin_amdgcn_wmma_f32_16x16x32_bf16(false, a0, false, b1, (short)0, c01, false, false);
        c10 = __builtin_amdgcn_wmma_f32_16x16x32_bf16(false, a1, false, b0, (short)0, c10, false, false);
        c11 = __builtin_amdgcn_wmma_f32_16x16x32_bf16(false, a1, false, b1, (short)0, c11, false, false);

        if (more) {
            commitB(p ^ 1);                     // waits for globals *after* WMMAs
#if HAS_ASYNC_LDS
            WAIT_ASYNC();                       // drain A DMA before barrier signal
#else
            commitA(p ^ 1);
#endif
        }
        p ^= 1;
    }

    // --- epilogue: C/D layout = VGPR r holds M=r (lanes 0-15) / M=8+r (lanes 16-31)
    const int m0 = cb + mb + ((lane >> 4) << 3);
    const int n0 = nb_ + (lane & 15);
    const long rowOff = (long)h * WW;
#pragma unroll
    for (int r = 0; r < 8; ++r) {
        int coA = m0 + r;        // a0 tiles
        int coB = m0 + 16 + r;   // a1 tiles
        float bvA = bias[coA];
        float bvB = bias[coB];
        long oA = ((long)(b * CO + coA)) * HWP + rowOff;
        long oB = ((long)(b * CO + coB)) * HWP + rowOff;
        out[oA + n0]      = c00[r] + bvA;
        out[oA + n0 + 16] = c01[r] + bvA;
        out[oB + n0]      = c10[r] + bvB;
        out[oB + n0 + 16] = c11[r] + bvB;
    }
}

extern "C" void kernel_launch(void* const* d_in, const int* in_sizes, int n_in,
                              void* d_out, int out_size, void* d_ws, size_t ws_size,
                              hipStream_t stream) {
    const float* x      = (const float*)d_in[0];
    const float* style  = (const float*)d_in[1];
    const float* weight = (const float*)d_in[2];
    const float* mod_w  = (const float*)d_in[3];
    const float* mod_b  = (const float*)d_in[4];
    const float* bias   = (const float*)d_in[5];
    float* out = (float*)d_out;

    // workspace layout
    float*  s     = (float*)d_ws;                    // NB*CI
    float*  demod = s + NB * CI;                     // NB*CO
    float*  wsum  = demod + NB * CO;                 // CO*CI
    __bf16* wb    = (__bf16*)(wsum + CO * CI);       // NB*CO*KT bf16 (~18.9 MB)

    wsum_kernel <<<(CO * CI) / 256, 256, 0, stream>>>(weight, wsum);
    style_kernel<<<(NB * CI) / 256, 256, 0, stream>>>(style, mod_w, mod_b, s);
    demod_kernel<<<(NB * CO) / 256, 256, 0, stream>>>(s, wsum, demod);
    prep_kernel <<<(NB * CO * KT) / 256, 256, 0, stream>>>(weight, s, demod, wb);

    dim3 grid(HH, CO / 64, NB);
    conv_kernel<<<grid, 256, 0, stream>>>(x, wb, bias, out);
}
